// Block_84121229459839
// MI455X (gfx1250) — matrix operand order repaired and checked by reading, not compiled, not run on hardware
//
#include <hip/hip_runtime.h>
#include <hip/hip_bf16.h>
#include <math.h>

// ---------------------------------------------------------------------------
// Types for WMMA fragments (gfx1250, wave32)
// ---------------------------------------------------------------------------
typedef __attribute__((ext_vector_type(16))) _Float16 v16h;
typedef __attribute__((ext_vector_type(8)))  _Float16 v8h;
typedef __attribute__((ext_vector_type(8)))  float    v8f;
typedef __attribute__((ext_vector_type(4)))  unsigned int v4u;
typedef __attribute__((ext_vector_type(8)))  int      v8i;
typedef __attribute__((ext_vector_type(4)))  int      v4i;

union Frag {
    v16h v;
    v8h  h[2];
};

__device__ inline v8f wmma_f16(const Frag& a, const Frag& b, v8f c) {
    return __builtin_amdgcn_wmma_f32_16x16x32_f16(
        /*neg_a=*/false, a.v, /*neg_b=*/false, b.v,
        /*c_mod=*/(short)0, c, /*reuse_a=*/false, /*reuse_b=*/false);
}

__device__ inline float quant_relu_f(float x, float alpha) {
    // relu -> x/alpha -> clip[0,1] -> round(*63)/63 -> *alpha   (RNE, like jnp.round)
    float xd = fmaxf(x, 0.0f) * (1.0f / alpha);
    xd = fminf(xd, 1.0f);
    return rintf(xd * 63.0f) * (1.0f / 63.0f) * alpha;
}

#if __has_builtin(__builtin_amdgcn_tensor_load_to_lds)
#define USE_TDM 1
#else
#define USE_TDM 0
#endif

// LDS byte offset of a __shared__ object: flat LDS addresses carry the LDS
// offset in addr[31:0] (ISA 10.2 aperture mapping), so truncation is exact.
__device__ inline unsigned lds_off_of(const void* p) {
    return (unsigned)(unsigned long long)(uintptr_t)p;
}

#if USE_TDM
// Issue a 2D TDM tile load: rows x kc f16 elements, row stride strideElems,
// packed row-major into LDS at lds_off. D# built per ISA 8.3/8.4.
// clang-23 toolchain: 6-arg builtin (g0 v4u, g1 v8i, g2 v4i, g3 v4i, v8i, cpol).
__device__ inline void tdm_load_2d_f16(unsigned lds_off, const void* gptr,
                                       unsigned rows, unsigned kc,
                                       unsigned long long strideElems) {
    unsigned long long ga = (unsigned long long)(uintptr_t)gptr;
    v4u g0;
    g0[0] = 1u;                                          // count=1, user mode
    g0[1] = lds_off;                                     // lds_addr [63:32]
    g0[2] = (unsigned)ga;                                // global_addr lo
    g0[3] = ((unsigned)(ga >> 32) & 0x01FFFFFFu) | (2u << 30); // ga[56:32] | type=2
    v8i g1;
    g1[0] = (int)(1u << 16);                             // data_size=1 (2 bytes)
    g1[1] = (int)((kc & 0xFFFFu) << 16);                 // tensor_dim0[15:0]
    g1[2] = (int)(((kc >> 16) & 0xFFFFu) | ((rows & 0xFFFFu) << 16)); // td0 hi | td1 lo
    g1[3] = (int)(((rows >> 16) & 0xFFFFu) | ((kc & 0xFFFFu) << 16)); // td1 hi | tile_dim0
    g1[4] = (int)(rows & 0xFFFFu);                       // tile_dim1 (tile_dim2=0)
    g1[5] = (int)(unsigned)(strideElems & 0xFFFFFFFFull);// tensor_dim0_stride lo
    g1[6] = (int)(unsigned)((strideElems >> 32) & 0xFFFFull); // stride hi (td1_stride=0)
    g1[7] = 0;
    v4i z4 = {0, 0, 0, 0};
    v8i z8 = {0, 0, 0, 0, 0, 0, 0, 0};
    __builtin_amdgcn_tensor_load_to_lds(g0, g1, z4, z4, z8, 0);
}
#endif

// ---------------------------------------------------------------------------
// fp32 -> f16 conversion (weights)
// ---------------------------------------------------------------------------
__global__ void cvt_f32_f16_kernel(const float* __restrict__ in,
                                   _Float16* __restrict__ out, size_t n) {
    size_t i = (size_t)blockIdx.x * blockDim.x + threadIdx.x;
    if (i < n) out[i] = (_Float16)in[i];
}

// ---------------------------------------------------------------------------
// Fused RMSNorm + quant_relu, one row (C=2048) per block, f16 output
// ---------------------------------------------------------------------------
__global__ __launch_bounds__(256)
void rmsnorm_quant_kernel(const float* __restrict__ x,
                          const float* __restrict__ w,
                          const float* __restrict__ alpha_p,
                          _Float16* __restrict__ out, int C) {
    __shared__ float red[8];
    const int t = blockIdx.x;
    const float* xr = x + (size_t)t * C;
    float ss = 0.0f;
    for (int i = threadIdx.x; i < C; i += blockDim.x) {
        float v = xr[i];
        ss += v * v;
    }
    for (int o = 16; o >= 1; o >>= 1) ss += __shfl_xor(ss, o, 32);
    if ((threadIdx.x & 31) == 0) red[threadIdx.x >> 5] = ss;
    __syncthreads();
    if (threadIdx.x < 32) {
        float v = (threadIdx.x < 8) ? red[threadIdx.x] : 0.0f;
        for (int o = 4; o >= 1; o >>= 1) v += __shfl_xor(v, o, 32);
        if (threadIdx.x == 0) red[0] = v;
    }
    __syncthreads();
    const float inv = rsqrtf(red[0] / (float)C + 1e-5f);
    const float alpha = alpha_p[0];
    _Float16* orow = out + (size_t)t * C;
    for (int i = threadIdx.x; i < C; i += blockDim.x) {
        float nv = w[i] * xr[i] * inv;
        orow[i] = (_Float16)quant_relu_f(nv, alpha);
    }
}

// ---------------------------------------------------------------------------
// Elementwise epilogues
// ---------------------------------------------------------------------------
__global__ void quant_to_f16_kernel(const float* __restrict__ in,
                                    const float* __restrict__ alpha_p,
                                    _Float16* __restrict__ out, size_t n) {
    size_t i = (size_t)blockIdx.x * blockDim.x + threadIdx.x;
    if (i < n) out[i] = (_Float16)quant_relu_f(in[i], alpha_p[0]);
}

__global__ void silu_mul_kernel(const float* __restrict__ gate,
                                const float* __restrict__ up,
                                _Float16* __restrict__ out, size_t n) {
    size_t i = (size_t)blockIdx.x * blockDim.x + threadIdx.x;
    if (i < n) {
        float g = gate[i];
        float s = g / (1.0f + __expf(-g));
        out[i] = (_Float16)(s * up[i]);
    }
}

__global__ void add_f32_kernel(const float* __restrict__ a,
                               const float* __restrict__ b,
                               float* __restrict__ out, size_t n) {
    size_t i = (size_t)blockIdx.x * blockDim.x + threadIdx.x;
    if (i < n) out[i] = a[i] + b[i];
}

// Repack V transposed per kv-group: vt[(g*64+hs)*T + t] = qkv16[t*3072 + g*384 + 320 + hs]
__global__ void build_vt_kernel(const _Float16* __restrict__ qkv16,
                                _Float16* __restrict__ vt, int T) {
    size_t i = (size_t)blockIdx.x * blockDim.x + threadIdx.x;
    size_t total = (size_t)8 * 64 * T;
    if (i >= total) return;
    int t  = (int)(i % T);
    int hs = (int)((i / T) % 64);
    int g  = (int)(i / ((size_t)T * 64));
    vt[i] = qkv16[(size_t)t * 3072 + g * 384 + 320 + hs];
}

// ---------------------------------------------------------------------------
// WMMA GEMM with TDM-staged LDS double buffering:
//   C[M,N](f32) = A[M,K](f16) * B[N,K](f16)^T
// Block = 256 threads (8 waves: 2 in M x 4 in N). Block tile 64(M) x 128(N).
// K staged in KC=64 chunks: A 64xKC (8KB) + B 128xKC (16KB), double buffered
// (48KB LDS). Wave 0 issues tensor_load_to_lds for chunk c+1 while all waves
// compute chunk c from LDS; sync via s_wait_tensorcnt + workgroup barriers.
// Fragment layouts (ISA 7.12.2):
//   A 16x32: lane m=lane&15, K chunks [kb,kb+8)+[kb+16,kb+24), kb=(lane>>4)*8
//   B 32x16: lane n=lane&15, K chunks [kB,kB+8)+[kB+16,kB+24), kB=(lane>>4)*8
// ---------------------------------------------------------------------------
#define KC 64

__global__ __launch_bounds__(256)
void gemm_wmma_tdm_kernel(const _Float16* __restrict__ A,
                          const _Float16* __restrict__ B,
                          float* __restrict__ C,
                          int M, int N, int K) {
    __shared__ _Float16 Abuf[2][64 * KC];
    __shared__ _Float16 Bbuf[2][128 * KC];

    const int wave  = threadIdx.x >> 5;
    const int lane  = threadIdx.x & 31;
    const int waveM = wave >> 2;        // 0..1
    const int waveN = wave & 3;         // 0..3
    const int mBlock = blockIdx.y * 64;
    const int nBlock = blockIdx.x * 128;

    const int mA  = lane & 15;
    const int kbA = (lane >> 4) * 8;
    const int nB  = lane & 15;
    const int kbB = (lane >> 4) * 8;

    const int nChunks = K / KC;

#if USE_TDM
    auto stage = [&](int chunk, int buf) {
        if (threadIdx.x < 32) {  // one TDM issue per instruction; wave 0 only
            const int k0 = chunk * KC;
            tdm_load_2d_f16(lds_off_of(&Abuf[buf][0]),
                            A + (size_t)mBlock * K + k0, 64, KC,
                            (unsigned long long)K);
            tdm_load_2d_f16(lds_off_of(&Bbuf[buf][0]),
                            B + (size_t)nBlock * K + k0, 128, KC,
                            (unsigned long long)K);
        }
    };
    stage(0, 0);
    if (threadIdx.x < 32) __builtin_amdgcn_s_wait_tensorcnt(0);
    __syncthreads();
#else
    auto stage_sync = [&](int chunk, int buf) {
        const int k0 = chunk * KC;
        for (int idx = threadIdx.x; idx < 64 * (KC / 8); idx += 256) {
            int r = idx >> 3, c = (idx & 7) * 8;
            *(v8h*)&Abuf[buf][r * KC + c] =
                *(const v8h*)(A + (size_t)(mBlock + r) * K + k0 + c);
        }
        for (int idx = threadIdx.x; idx < 128 * (KC / 8); idx += 256) {
            int r = idx >> 3, c = (idx & 7) * 8;
            *(v8h*)&Bbuf[buf][r * KC + c] =
                *(const v8h*)(B + (size_t)(nBlock + r) * K + k0 + c);
        }
    };
    stage_sync(0, 0);
    __syncthreads();
#endif

    v8f acc00 = {}, acc01 = {}, acc10 = {}, acc11 = {};

    for (int c = 0; c < nChunks; ++c) {
        const int buf = c & 1;
#if USE_TDM
        if (c + 1 < nChunks) stage(c + 1, buf ^ 1);
#endif
        const _Float16* Ab0 = &Abuf[buf][(waveM * 32 + mA) * KC + kbA];
        const _Float16* Ab1 = Ab0 + 16 * KC;
        const _Float16* Bb0 = &Bbuf[buf][(waveN * 32 + nB) * KC + kbB];
        const _Float16* Bb1 = Bb0 + 16 * KC;
#pragma unroll
        for (int k0 = 0; k0 < KC; k0 += 32) {
            Frag a0, a1, b0, b1;
            a0.h[0] = *(const v8h*)(Ab0 + k0);
            a0.h[1] = *(const v8h*)(Ab0 + k0 + 16);
            a1.h[0] = *(const v8h*)(Ab1 + k0);
            a1.h[1] = *(const v8h*)(Ab1 + k0 + 16);
            b0.h[0] = *(const v8h*)(Bb0 + k0);
            b0.h[1] = *(const v8h*)(Bb0 + k0 + 16);
            b1.h[0] = *(const v8h*)(Bb1 + k0);
            b1.h[1] = *(const v8h*)(Bb1 + k0 + 16);
            acc00 = wmma_f16(a0, b0, acc00);
            acc01 = wmma_f16(a0, b1, acc01);
            acc10 = wmma_f16(a1, b0, acc10);
            acc11 = wmma_f16(a1, b1, acc11);
        }
        __syncthreads();             // all waves done reading buf
#if USE_TDM
        if (threadIdx.x < 32 && c + 1 < nChunks)
            __builtin_amdgcn_s_wait_tensorcnt(0);   // buf^1 fully written
        __syncthreads();             // release consumers
#else
        if (c + 1 < nChunks) stage_sync(c + 1, buf ^ 1);
        __syncthreads();
#endif
    }

    // C/D layout: VGPR r -> M = r + 8*(lane>>4), N = lane&15
    const int ms  = (lane >> 4) * 8;
    const int col = lane & 15;
    const int m0 = mBlock + waveM * 32;
    const int n0 = nBlock + waveN * 32;
    float* C00 = C + (size_t)(m0 + ms) * N + n0 + col;
#pragma unroll
    for (int r = 0; r < 8; ++r) {
        C00[(size_t)r * N]             = acc00[r];
        C00[(size_t)r * N + 16]        = acc01[r];
        C00[((size_t)r + 16) * N]      = acc10[r];
        C00[((size_t)r + 16) * N + 16] = acc11[r];
    }
}

// ---------------------------------------------------------------------------
// Attention: one block per (16-query tile, head). 256 threads = 8 waves.
// Dynamic LDS: S[16][T] f32 scores (128 KB) + 4*16*16 f32 split-K scratch.
// Phase 1: S = mask(QK^T * 1/8) via WMMA (8 waves over 128 key tiles)
// Phase 2: exact softmax per row + quant_relu(alpha_sm); quantized P packed
//          as f16 in place over the front half of each f32 score row (row is
//          wave-private; LDS ops from one wave are in-order, loads precede
//          stores per iteration -> no hazard).
// Phase 3: Y = P*V via WMMA (f16 fragments straight from LDS), split-K over
//          wave pairs, combined through LDS.
// ---------------------------------------------------------------------------
__global__ __launch_bounds__(256)
void attention_kernel(const _Float16* __restrict__ qkv16,
                      const _Float16* __restrict__ vt,
                      const float* __restrict__ alpha_sm_p,
                      _Float16* __restrict__ y, int T) {
    extern __shared__ float smem[];
    float* S   = smem;              // 16 * T floats
    float* red = smem + 16 * T;     // 4 * 256 floats

    const int h  = blockIdx.y;
    const int m0 = blockIdx.x * 16;
    const int g  = h >> 2;          // kv group
    const int qi = h & 3;           // q head within group
    const int wave = threadIdx.x >> 5;
    const int lane = threadIdx.x & 31;
    const int rowA = lane & 15;
    const int kbA  = (lane >> 4) * 8;
    const int nB   = lane & 15;
    const int kbB  = (lane >> 4) * 8;
    const float alpha = alpha_sm_p[0];
    const float NEG = -3.0e38f;

    // --- Phase 1: scores ---
    const _Float16* Qp = qkv16 + (size_t)(m0 + rowA) * 3072 + g * 384 + qi * 64 + kbA;
    Frag qa0, qa1;
    qa0.h[0] = *(const v8h*)(Qp);
    qa0.h[1] = *(const v8h*)(Qp + 16);
    qa1.h[0] = *(const v8h*)(Qp + 32);
    qa1.h[1] = *(const v8h*)(Qp + 48);

    const int nTiles = T / 16;
    for (int kt = wave; kt < nTiles; kt += 8) {
        const int key0 = kt * 16;
        const _Float16* Kp = qkv16 + (size_t)(key0 + nB) * 3072 + g * 384 + 256 + kbB;
        Frag kb0, kb1;
        kb0.h[0] = *(const v8h*)(Kp);
        kb0.h[1] = *(const v8h*)(Kp + 16);
        kb1.h[0] = *(const v8h*)(Kp + 32);
        kb1.h[1] = *(const v8h*)(Kp + 48);
        v8f acc = {};
        acc = wmma_f16(qa0, kb0, acc);
        acc = wmma_f16(qa1, kb1, acc);

        const int ms  = (lane >> 4) * 8;
        const int col = lane & 15;
#pragma unroll
        for (int r = 0; r < 8; ++r) {
            int m = ms + r;
            int key = key0 + col;
            float v = acc[r] * 0.125f;            // 1/sqrt(64)
            if (key > m0 + m) v = NEG;            // causal mask
            S[m * T + key] = v;
        }
    }
    __syncthreads();

    // --- Phase 2: softmax + quant per row (wave -> 2 rows), pack f16 ---
    for (int rr = 0; rr < 2; ++rr) {
        const int m = wave * 2 + rr;
        float* Sr = S + m * T;
        _Float16* Ph = (_Float16*)Sr;             // in-place f16 pack
        float mx = NEG;
        for (int i = lane; i < T; i += 32) mx = fmaxf(mx, Sr[i]);
        for (int o = 16; o >= 1; o >>= 1) mx = fmaxf(mx, __shfl_xor(mx, o, 32));
        float sum = 0.0f;
        for (int i = lane; i < T; i += 32) sum += __expf(Sr[i] - mx);
        for (int o = 16; o >= 1; o >>= 1) sum += __shfl_xor(sum, o, 32);
        const float invs = 1.0f / sum;
        for (int i = lane; i < T; i += 32) {
            float p = __expf(Sr[i] - mx) * invs;
            Ph[i] = (_Float16)quant_relu_f(p, alpha);
        }
    }
    __syncthreads();

    // --- Phase 3: Y = P * V, split keys across wave pairs ---
    const _Float16* Sh = (const _Float16*)S;      // row m at Sh + m*2*T
    const int nt    = wave & 3;       // hs tile 0..3
    const int khalf = wave >> 2;      // key half 0/1
    const int koff  = khalf * (T / 2);
    v8f acc = {};
    const _Float16* Vp = vt + (size_t)(g * 64 + nt * 16 + nB) * T + koff + kbB;
    const _Float16* Pp = Sh + rowA * 2 * T + koff + kbA;

    for (int k0 = 0; k0 < T / 2; k0 += 32) {
        Frag pa, vb;
        pa.h[0] = *(const v8h*)(Pp + k0);
        pa.h[1] = *(const v8h*)(Pp + k0 + 16);
        vb.h[0] = *(const v8h*)(Vp + k0);
        vb.h[1] = *(const v8h*)(Vp + k0 + 16);
        acc = wmma_f16(pa, vb, acc);
    }

    const int ms  = (lane >> 4) * 8;
    const int col = lane & 15;
    if (khalf == 1) {
        float* dst = red + nt * 256;
#pragma unroll
        for (int r = 0; r < 8; ++r) dst[(ms + r) * 16 + col] = acc[r];
    }
    __syncthreads();
    if (khalf == 0) {
        const float* src = red + nt * 256;
#pragma unroll
        for (int r = 0; r < 8; ++r) {
            float v = acc[r] + src[(ms + r) * 16 + col];
            y[(size_t)(m0 + ms + r) * 2048 + h * 64 + nt * 16 + col] = (_Float16)v;
        }
    }
}

// ---------------------------------------------------------------------------
// Host-side orchestration
// ---------------------------------------------------------------------------
static inline size_t align256(size_t x) { return (x + 255) & ~(size_t)255; }

extern "C" void kernel_launch(void* const* d_in, const int* in_sizes, int n_in,
                              void* d_out, int out_size, void* d_ws, size_t ws_size,
                              hipStream_t stream) {
    (void)in_sizes; (void)n_in; (void)out_size; (void)ws_size;
    constexpr int T = 2048, C = 2048, FF = 5632;
    constexpr int O = 3072;              // (H + 2G) * HS
    constexpr size_t TC  = (size_t)T * C;
    constexpr size_t TO  = (size_t)T * O;
    constexpr size_t TFF = (size_t)T * FF;

    const float* x        = (const float*)d_in[0];
    const float* w_rms1   = (const float*)d_in[1];
    const float* w_rms2   = (const float*)d_in[2];
    const float* alpha1   = (const float*)d_in[3];
    const float* alpha2   = (const float*)d_in[4];
    const float* alpha_q  = (const float*)d_in[5];
    const float* alpha_sm = (const float*)d_in[6];
    const float* attn_w   = (const float*)d_in[7];
    const float* proj_w   = (const float*)d_in[8];
    const float* fc1_w    = (const float*)d_in[9];
    const float* fc2_w    = (const float*)d_in[10];
    const float* mproj_w  = (const float*)d_in[11];
    float* out = (float*)d_out;

    char* ws = (char*)d_ws;
    size_t off = 0;
    auto take = [&](size_t bytes) { char* p = ws + off; off = align256(off + bytes); return p; };

    _Float16* wAttn16  = (_Float16*)take((size_t)O  * C  * 2);
    _Float16* wProj16  = (_Float16*)take((size_t)C  * C  * 2);
    _Float16* wFc116   = (_Float16*)take((size_t)FF * C  * 2);
    _Float16* wFc216   = (_Float16*)take((size_t)FF * C  * 2);
    _Float16* wMproj16 = (_Float16*)take((size_t)C  * FF * 2);
    _Float16* n16      = (_Float16*)take(TC * 2);      // n1, later n2
    float*    bigF     = (float*)   take(TO * 4);      // qkv f32; reused for proj/mlp out
    _Float16* qkv16    = (_Float16*)take(TO * 2);
    _Float16* vt16     = (_Float16*)take((size_t)8 * 64 * T * 2);
    _Float16* y16      = (_Float16*)take(TC * 2);
    float*    gateF    = (float*)   take(TFF * 4);
    float*    upF      = (float*)   take(TFF * 4);
    float*    x2       = (float*)   take(TC * 4);
    _Float16* h16      = (_Float16*)take(TFF * 2);

    auto eltGrid = [](size_t n) { return dim3((unsigned)((n + 255) / 256)); };

    // 1) Weights fp32 -> f16 (resident in the 192MB L2 afterwards)
    cvt_f32_f16_kernel<<<eltGrid((size_t)O * C),  256, 0, stream>>>(attn_w,  wAttn16,  (size_t)O * C);
    cvt_f32_f16_kernel<<<eltGrid((size_t)C * C),  256, 0, stream>>>(proj_w,  wProj16,  (size_t)C * C);
    cvt_f32_f16_kernel<<<eltGrid((size_t)FF * C), 256, 0, stream>>>(fc1_w,   wFc116,   (size_t)FF * C);
    cvt_f32_f16_kernel<<<eltGrid((size_t)FF * C), 256, 0, stream>>>(fc2_w,   wFc216,   (size_t)FF * C);
    cvt_f32_f16_kernel<<<eltGrid((size_t)C * FF), 256, 0, stream>>>(mproj_w, wMproj16, (size_t)C * FF);

    // 2) n1 = quant_relu(rmsnorm(x, w_rms1), alpha1)
    rmsnorm_quant_kernel<<<T, 256, 0, stream>>>(x, w_rms1, alpha1, n16, C);

    // 3) QKV GEMM: [T,O] = n1[T,C] x attn_w[O,C]^T
    gemm_wmma_tdm_kernel<<<dim3(O / 128, T / 64), 256, 0, stream>>>(n16, wAttn16, bigF, T, O, C);

    // 4) quant_relu(qkv, alpha_q) -> f16
    quant_to_f16_kernel<<<eltGrid(TO), 256, 0, stream>>>(bigF, alpha_q, qkv16, TO);

    // 5) Transposed V repack for P*V B-fragments
    build_vt_kernel<<<eltGrid((size_t)8 * 64 * T), 256, 0, stream>>>(qkv16, vt16, T);

    // 6) Attention
    {
        size_t smem = (size_t)16 * T * 4 + 4 * 256 * 4;
        attention_kernel<<<dim3(T / 16, 32), 256, smem, stream>>>(qkv16, vt16, alpha_sm, y16, T);
    }

    // 7) Output projection: [T,C] = y[T,C] x proj_w[C,C]^T   (reuse bigF)
    gemm_wmma_tdm_kernel<<<dim3(C / 128, T / 64), 256, 0, stream>>>(y16, wProj16, bigF, T, C, C);

    // 8) x2 = x + attn_out
    add_f32_kernel<<<eltGrid(TC), 256, 0, stream>>>(x, bigF, x2, TC);

    // 9) n2 = quant_relu(rmsnorm(x2, w_rms2), alpha2)
    rmsnorm_quant_kernel<<<T, 256, 0, stream>>>(x2, w_rms2, alpha2, n16, C);

    // 10/11) gate = n2 x fc1^T ; up = n2 x fc2^T
    gemm_wmma_tdm_kernel<<<dim3(FF / 128, T / 64), 256, 0, stream>>>(n16, wFc116, gateF, T, FF, C);
    gemm_wmma_tdm_kernel<<<dim3(FF / 128, T / 64), 256, 0, stream>>>(n16, wFc216, upF,   T, FF, C);

    // 12) h = silu(gate) * up -> f16
    silu_mul_kernel<<<eltGrid(TFF), 256, 0, stream>>>(gateF, upF, h16, TFF);

    // 13) mlp_out: [T,C] = h[T,FF] x mlp_proj_w[C,FF]^T   (reuse bigF)
    gemm_wmma_tdm_kernel<<<dim3(C / 128, T / 64), 256, 0, stream>>>(h16, wMproj16, bigF, T, C, FF);

    // 14) out = x2 + mlp_out
    add_f32_kernel<<<eltGrid(TC), 256, 0, stream>>>(x2, bigF, out, TC);
}
